// Generator_20083267076515
// MI455X (gfx1250) — matrix operand order repaired and checked by reading, not compiled, not run on hardware
//
#include <hip/hip_runtime.h>
#include <math.h>

// ---------------------------------------------------------------------------
// Problem constants (from the reference)
// ---------------------------------------------------------------------------
#define Bn      16
#define Tn      400
#define Hn      400
#define Vn      22000
#define Sn      30
#define STEPSn  8
#define NGn     3
#define H3      (3 * Hn)          // 1200
#define KP      416               // H padded to multiple of 32 (WMMA K=32)
#define TP      416               // T padded to multiple of 32
#define MROWS   (Bn * Sn * STEPSn) // 3840 decoder states; row m=(b*S+s)*STEPS+t
#define PAIRS   (Bn * Sn)         // 480 independent GRU chains
#define KSTEPS  (KP / 32)         // 13
#define VBLK    128               // vocab columns per block (8 waves x 16)
#define NVB     ((Vn + VBLK - 1) / VBLK)  // 172

typedef __bf16 bf16;
typedef __attribute__((ext_vector_type(16))) bf16  v16bf;
typedef __attribute__((ext_vector_type(8)))  bf16  v8bf;
typedef __attribute__((ext_vector_type(8)))  float v8f;

// ---------------------------------------------------------------------------
// WMMA helpers (CDNA5 wave32, V_WMMA_F32_16X16X32_BF16)
//
// A (16x32 MxK, bf16, row-major): lane L<16 -> row L, K={k0..k0+7,k0+16..k0+23}
//                                 lane L>=16 -> row L-16, K={k0+8..15,k0+24..31}
//   => two contiguous 16B (v8bf) loads per lane.
// B (32x16 KxN, bf16): stored PANELIZED: Bp[((k/32)*N + n)*32 + (k%32)].
//   lane L<16 -> col n0+L, K={k0..k0+7,k0+16..k0+23}; lane L>=16 -> col n0+L-16, K={k0+8..15,k0+24..31}
//   => two contiguous 16B loads per lane (same as A).
// C/D (16x16 f32): VGPR g, lane l -> row g + 8*(l>=16), col l&15.
// ---------------------------------------------------------------------------
__device__ __forceinline__ v8f wmma_bf16(v16bf a, v16bf b, v8f c) {
  return __builtin_amdgcn_wmma_f32_16x16x32_bf16(false, a, false, b,
                                                 (short)0, c, false, false);
}

__device__ __forceinline__ v16bf load_a16x32(const bf16* base, int ld, int k0,
                                             int lane) {
  int r  = lane & 15;
  int ko = k0 + ((lane & 16) ? 8 : 0);
  const bf16* p = base + (size_t)r * ld + ko;
  v8bf lo = *(const v8bf*)p;
  v8bf hi = *(const v8bf*)(p + 16);
  return __builtin_shufflevector(lo, hi, 0, 1, 2, 3, 4, 5, 6, 7,
                                 8, 9, 10, 11, 12, 13, 14, 15);
}

// Panelized B load: two contiguous 16B loads per lane.
__device__ __forceinline__ v16bf load_b_panel(const bf16* P, int N, int k0,
                                              int n0, int lane) {
  int col = n0 + (lane & 15);
  const bf16* p = P + ((size_t)(k0 >> 5) * N + col) * 32 + ((lane & 16) ? 8 : 0);
  v8bf lo = *(const v8bf*)p;
  v8bf hi = *(const v8bf*)(p + 16);
  return __builtin_shufflevector(lo, hi, 0, 1, 2, 3, 4, 5, 6, 7,
                                 8, 9, 10, 11, 12, 13, 14, 15);
}

__device__ __forceinline__ float sigmoidf_(float x) {
  return 1.0f / (1.0f + __expf(-x));
}

// ---------------------------------------------------------------------------
// Prep kernels: bf16 conversion into panelized-B layouts
// ---------------------------------------------------------------------------
// dst panel[((k/32)*N + n)*32 + k%32] = src[n][k] (f32 [N][K]); zero pad k>=K
__global__ void k_panelcvt(bf16* __restrict__ dst, const float* __restrict__ src,
                           int N, int K) {
  long total = (long)KP * N;     // KP = padded K (multiple of 32)
  for (long id = (long)blockIdx.x * blockDim.x + threadIdx.x; id < total;
       id += (long)gridDim.x * blockDim.x) {
    int ki = (int)(id & 31);
    long rest = id >> 5;
    int n  = (int)(rest % N);
    int kb = (int)(rest / N);
    int k  = kb * 32 + ki;
    dst[id] = (k < K) ? (bf16)src[(size_t)n * K + k] : (bf16)0.0f;
  }
}

// encBp[b]: panel over K=t of enc[b][t][h]  (context GEMM B, N=Hn, K=TP)
__global__ void k_panel_encB(bf16* __restrict__ dst, const float* __restrict__ src) {
  long per = (long)TP * Hn;
  long total = (long)Bn * per;
  for (long id = (long)blockIdx.x * blockDim.x + threadIdx.x; id < total;
       id += (long)gridDim.x * blockDim.x) {
    int b = (int)(id / per);
    long r = id % per;
    int ti = (int)(r & 31);
    long rest = r >> 5;
    int h  = (int)(rest % Hn);
    int tb = (int)(rest / Hn);
    int t  = tb * 32 + ti;
    dst[id] = (t < Tn) ? (bf16)src[((size_t)b * Tn + t) * Hn + h] : (bf16)0.0f;
  }
}

// encTp[b]: panel over K=h of enc[b][t][h]^T (attention GEMM B, N=Tn, K=KP)
__global__ void k_panel_encT(bf16* __restrict__ dst, const float* __restrict__ src) {
  long per = (long)KP * Tn;
  long total = (long)Bn * per;
  for (long id = (long)blockIdx.x * blockDim.x + threadIdx.x; id < total;
       id += (long)gridDim.x * blockDim.x) {
    int b = (int)(id / per);
    long r = id % per;
    int ki = (int)(r & 31);
    long rest = r >> 5;
    int t  = (int)(rest % Tn);
    int kb = (int)(rest / Tn);
    int k  = kb * 32 + ki;
    dst[id] = (k < Hn) ? (bf16)src[((size_t)b * Tn + t) * Hn + k] : (bf16)0.0f;
  }
}

// Teacher-forced decoder inputs known up front:
//   x(t=0) = slot_emb[s];  x(t>0) = embedding[target[b][s][t-1]]
__global__ void k_build_x(float* __restrict__ xF, bf16* __restrict__ xB,
                          const float* __restrict__ slot_embs,
                          const float* __restrict__ emb,
                          const int* __restrict__ target) {
  long total = (long)MROWS * KP;
  for (long id = (long)blockIdx.x * blockDim.x + threadIdx.x; id < total;
       id += (long)gridDim.x * blockDim.x) {
    int m = (int)(id / KP), k = (int)(id % KP);
    int b = m / (Sn * STEPSn), rem = m % (Sn * STEPSn);
    int s = rem / STEPSn, t = rem % STEPSn;
    float v = 0.0f;
    if (k < Hn) {
      if (t == 0) v = slot_embs[(size_t)s * Hn + k];
      else {
        int w = target[((size_t)b * Sn + s) * STEPSn + (t - 1)];
        v = emb[(size_t)w * Hn + k];
      }
      xF[(size_t)m * Hn + k] = v;
    }
    xB[id] = (bf16)v;
  }
}

__global__ void k_zero_bf16(bf16* __restrict__ p, long n) {
  for (long id = (long)blockIdx.x * blockDim.x + threadIdx.x; id < n;
       id += (long)gridDim.x * blockDim.x)
    p[id] = (bf16)0.0f;
}

// ---------------------------------------------------------------------------
// GRU step t (sequential over t, parallel over 480 (b,s) chains):
// gi = X_t @ Wih^T, gh = H_{t-1} @ Whh^T  (WMMA bf16 -> f32), then gates.
// Each wave owns a 16-col tile j0 and accumulates the matching columns of all
// three gate groups (r,z,n) for gi and gh -> elementwise update in registers.
// grid = PAIRS/16 = 30 blocks x 256 threads (8 waves).
// ---------------------------------------------------------------------------
__global__ void k_gru_step(const bf16* __restrict__ xB,
                           bf16* __restrict__ hB, float* __restrict__ hF,
                           const bf16* __restrict__ WihP,
                           const bf16* __restrict__ WhhP,
                           const float* __restrict__ bih,
                           const float* __restrict__ bhh,
                           const float* __restrict__ enc_hidden, int t) {
  __shared__ bf16 lds_x[16 * KP];
  __shared__ bf16 lds_h[16 * KP];
  int p0 = blockIdx.x * 16;
  int tid = threadIdx.x;
  for (int idx = tid; idx < 16 * KP; idx += blockDim.x) {
    int r = idx / KP, k = idx % KP;
    int p = p0 + r, m = p * STEPSn + t;
    lds_x[idx] = xB[(size_t)m * KP + k];
    if (t == 0)
      lds_h[idx] = (k < Hn) ? (bf16)enc_hidden[(size_t)(p / Sn) * Hn + k]
                            : (bf16)0.0f;
    else
      lds_h[idx] = hB[(size_t)(m - 1) * KP + k];
  }
  __syncthreads();

  int wave = tid >> 5, lane = tid & 31;
  for (int it = 0; it < 4; ++it) {
    int ct = wave + it * 8;
    if (ct >= Hn / 16) continue;   // 25 column tiles; no barriers below
    int j0 = ct * 16;
    v8f air = {}, aiz = {}, ain = {}, ahr = {}, ahz = {}, ahn = {};
#pragma unroll 1
    for (int ks = 0; ks < KSTEPS; ++ks) {
      int k0 = ks * 32;
      v16bf ax = load_a16x32(lds_x, KP, k0, lane);
      v16bf ah = load_a16x32(lds_h, KP, k0, lane);
      v16bf w;
      w = load_b_panel(WihP, H3, k0, j0,          lane); air = wmma_bf16(ax, w, air);
      w = load_b_panel(WihP, H3, k0, Hn + j0,     lane); aiz = wmma_bf16(ax, w, aiz);
      w = load_b_panel(WihP, H3, k0, 2 * Hn + j0, lane); ain = wmma_bf16(ax, w, ain);
      w = load_b_panel(WhhP, H3, k0, j0,          lane); ahr = wmma_bf16(ah, w, ahr);
      w = load_b_panel(WhhP, H3, k0, Hn + j0,     lane); ahz = wmma_bf16(ah, w, ahz);
      w = load_b_panel(WhhP, H3, k0, 2 * Hn + j0, lane); ahn = wmma_bf16(ah, w, ahn);
    }
    int j = j0 + (lane & 15);
    float bir = bih[j], biz = bih[Hn + j], bin = bih[2 * Hn + j];
    float bhr = bhh[j], bhz = bhh[Hn + j], bhn = bhh[2 * Hn + j];
#pragma unroll
    for (int g = 0; g < 8; ++g) {
      int r = g + ((lane & 16) ? 8 : 0);
      int p = p0 + r, m = p * STEPSn + t, b = p / Sn;
      float rr = sigmoidf_((air[g] + bir) + (ahr[g] + bhr));
      float zz = sigmoidf_((aiz[g] + biz) + (ahz[g] + bhz));
      float nn = tanhf((ain[g] + bin) + rr * (ahn[g] + bhn));
      float hp = (t == 0) ? enc_hidden[(size_t)b * Hn + j]
                          : hF[(size_t)(m - 1) * Hn + j];
      float hv = (1.0f - zz) * nn + zz * hp;
      hF[(size_t)m * Hn + j] = hv;
      hB[(size_t)m * KP + j] = (bf16)hv;
    }
  }
}

// ---------------------------------------------------------------------------
// Generic batched tile-GEMM: C[batch] = A[batch] @ B[batch], bf16 -> f32.
// A row-major (padded), B panelized. Used for attention logits and context.
// grid.x = batches * mtiles; 8 waves loop over N tiles.
// ---------------------------------------------------------------------------
__global__ void k_tile_gemm(const bf16* __restrict__ A, int lda, long strideA,
                            const bf16* __restrict__ Bp, int nB, long strideB,
                            float* __restrict__ C, int ldc, long strideC,
                            int mtilesPerBatch, int ntiles, int kpad) {
  int batch = blockIdx.x / mtilesPerBatch;
  int mt    = blockIdx.x % mtilesPerBatch;
  const bf16* Ab = A + (size_t)batch * strideA + (size_t)mt * 16 * lda;
  const bf16* Bb = Bp + (size_t)batch * strideB;
  float*      Cb = C + (size_t)batch * strideC;
  int wave = threadIdx.x >> 5, lane = threadIdx.x & 31;
  for (int nt = wave; nt < ntiles; nt += 8) {
    v8f acc = {};
    for (int k0 = 0; k0 < kpad; k0 += 32) {
      v16bf a = load_a16x32(Ab, lda, k0, lane);
      v16bf b = load_b_panel(Bb, nB, k0, nt * 16, lane);
      acc = wmma_bf16(a, b, acc);
    }
#pragma unroll
    for (int g = 0; g < 8; ++g) {
      int r = g + ((lane & 16) ? 8 : 0);
      Cb[(size_t)(mt * 16 + r) * ldc + nt * 16 + (lane & 15)] = acc[g];
    }
  }
}

// ---------------------------------------------------------------------------
// Masked softmax over T per decoder state; emit f32 attn (for scatter) and
// bf16 padded attn (A-matrix for the context GEMM). One block per row.
// ---------------------------------------------------------------------------
__global__ void k_attn_softmax(float* __restrict__ attn, bf16* __restrict__ attnB,
                               const int* __restrict__ lens) {
  __shared__ float red[128];
  int m = blockIdx.x, tid = threadIdx.x;
  int b = m / (Sn * STEPSn);
  int len = lens[b];
  float* row = attn + (size_t)m * Tn;

  float mx = -3.0e38f;
  for (int tpos = tid; tpos < Tn; tpos += blockDim.x) {
    float v = (tpos < len) ? row[tpos] : -1.0e9f;
    mx = fmaxf(mx, v);
  }
  red[tid] = mx; __syncthreads();
  for (int s = 64; s > 0; s >>= 1) {
    if (tid < s) red[tid] = fmaxf(red[tid], red[tid + s]);
    __syncthreads();
  }
  mx = red[0]; __syncthreads();

  float sm = 0.0f;
  for (int tpos = tid; tpos < Tn; tpos += blockDim.x) {
    float v = (tpos < len) ? row[tpos] : -1.0e9f;
    sm += __expf(v - mx);
  }
  red[tid] = sm; __syncthreads();
  for (int s = 64; s > 0; s >>= 1) {
    if (tid < s) red[tid] += red[tid + s];
    __syncthreads();
  }
  sm = red[0];
  float inv = 1.0f / sm;

  for (int tpos = tid; tpos < TP; tpos += blockDim.x) {
    float a = 0.0f;
    if (tpos < Tn) {
      float v = (tpos < len) ? row[tpos] : -1.0e9f;
      a = __expf(v - mx) * inv;
      row[tpos] = a;
    }
    attnB[(size_t)m * TP + tpos] = (bf16)a;
  }
}

// ---------------------------------------------------------------------------
// p_gen per state + gate head (uses step-0 context). One thread per state.
// ---------------------------------------------------------------------------
__global__ void k_pgen_gate(const float* __restrict__ hF,
                            const float* __restrict__ ctxF,
                            const float* __restrict__ xF,
                            const float* __restrict__ Wr, const float* __restrict__ br,
                            const float* __restrict__ Wg, const float* __restrict__ bg,
                            float* __restrict__ pgen, float* __restrict__ gateOut) {
  int m = blockIdx.x * blockDim.x + threadIdx.x;
  if (m >= MROWS) return;
  int b = m / (Sn * STEPSn), rem = m % (Sn * STEPSn);
  int s = rem / STEPSn, t = rem % STEPSn;
  float acc = br[0];
  const float* h = hF + (size_t)m * Hn;
  const float* c = ctxF + (size_t)m * Hn;
  const float* x = xF + (size_t)m * Hn;
  for (int k = 0; k < Hn; ++k) {
    acc = fmaf(Wr[k], h[k], acc);
    acc = fmaf(Wr[Hn + k], c[k], acc);
    acc = fmaf(Wr[2 * Hn + k], x[k], acc);
  }
  pgen[m] = sigmoidf_(acc);
  if (t == 0) {
    for (int g = 0; g < NGn; ++g) {
      float ga = bg[g];
      for (int k = 0; k < Hn; ++k) ga = fmaf(Wg[(size_t)g * Hn + k], c[k], ga);
      gateOut[((size_t)s * Bn + b) * NGn + g] = ga;
    }
  }
}

// ---------------------------------------------------------------------------
// Vocab GEMM, flash-style two-pass softmax. Each block owns 128 vocab cols,
// stages its panelized E^T slab (KP x 128 bf16 = 104KB) in dynamic LDS
// (320KB/WGP on CDNA5), then loops over all 240 row tiles, TWO tiles per
// iteration per wave (shared B fragment) with A double-buffered across k.
//   phase 0: per-block per-row (max, sumexp) partials
//   phase 1: recompute logits, write p_gen * softmax to d_out
// ---------------------------------------------------------------------------
__global__ void k_vocab(const bf16* __restrict__ hB, const bf16* __restrict__ EtP,
                        const float* __restrict__ pgen,
                        float* __restrict__ partMax, float* __restrict__ partSum,
                        const float* __restrict__ rowMax,
                        const float* __restrict__ rowSum,
                        float* __restrict__ dout, int phase) {
  extern __shared__ bf16 ldsE[];      // panel layout, N=VBLK: [(k/32)*VBLK + c][k%32]
  __shared__ float redM[8][16];
  __shared__ float redS[8][16];
  __shared__ float blkM[16];

  int nbase = blockIdx.x * VBLK;
  int tid = threadIdx.x;
  // Stage panel slab: contiguous per k-block in both src (EtP) and dst (LDS).
  for (int idx = tid; idx < KP * VBLK; idx += blockDim.x) {
    int ki = idx & 31;
    int rest = idx >> 5;
    int c  = rest % VBLK;
    int kb = rest / VBLK;
    int n  = nbase + c;
    ldsE[idx] = (n < Vn) ? EtP[((size_t)kb * Vn + n) * 32 + ki] : (bf16)0.0f;
  }
  __syncthreads();

  int wave = tid >> 5, lane = tid & 31;
  int n0 = wave * 16;
  bool active = (nbase + n0) < Vn;

  // Uniform (barrier-containing) per-tile row statistics for phase 0.
  auto rowstat = [&](const v8f& acc, int m0) {
    float vmax[8];
#pragma unroll
    for (int g = 0; g < 8; ++g) vmax[g] = active ? acc[g] : -3.0e38f;
    for (int d = 1; d < 16; d <<= 1)
#pragma unroll
      for (int g = 0; g < 8; ++g)
        vmax[g] = fmaxf(vmax[g], __shfl_xor(vmax[g], d, 32));
    if ((lane & 15) == 0)
#pragma unroll
      for (int g = 0; g < 8; ++g)
        redM[wave][g + ((lane & 16) ? 8 : 0)] = vmax[g];
    __syncthreads();
    if (tid < 16) {
      float mm = redM[0][tid];
      for (int w = 1; w < 8; ++w) mm = fmaxf(mm, redM[w][tid]);
      blkM[tid] = mm;
    }
    __syncthreads();
    float vsum[8];
#pragma unroll
    for (int g = 0; g < 8; ++g) {
      int r = g + ((lane & 16) ? 8 : 0);
      vsum[g] = active ? __expf(acc[g] - blkM[r]) : 0.0f;
    }
    for (int d = 1; d < 16; d <<= 1)
#pragma unroll
      for (int g = 0; g < 8; ++g) vsum[g] += __shfl_xor(vsum[g], d, 32);
    if ((lane & 15) == 0)
#pragma unroll
      for (int g = 0; g < 8; ++g)
        redS[wave][g + ((lane & 16) ? 8 : 0)] = vsum[g];
    __syncthreads();
    if (tid < 16) {
      float ss = 0.0f;
      for (int w = 0; w < 8; ++w) ss += redS[w][tid];
      partMax[(size_t)blockIdx.x * MROWS + m0 + tid] = blkM[tid];
      partSum[(size_t)blockIdx.x * MROWS + m0 + tid] = ss;
    }
    __syncthreads();
  };

  auto store_tile = [&](const v8f& acc, int m0) {
#pragma unroll
    for (int g = 0; g < 8; ++g) {
      int r = g + ((lane & 16) ? 8 : 0);
      int m = m0 + r;
      float scale = pgen[m] / rowSum[m];
      float Mx = rowMax[m];
      int b = m / (Sn * STEPSn), rem = m % (Sn * STEPSn);
      int s = rem / STEPSn, t = rem % STEPSn;
      size_t o = (((size_t)s * Bn + b) * STEPSn + t) * Vn
               + nbase + n0 + (lane & 15);
      dout[o] = scale * __expf(acc[g] - Mx);
    }
  };

  // Two row-tiles per iteration: one B fragment feeds two WMMAs; the A
  // fragments for k+32 are prefetched while the current WMMAs execute.
  for (int rt = 0; rt < MROWS / 16; rt += 2) {
    int m0 = rt * 16;
    v8f acc0 = {}, acc1 = {};
    if (active) {
      const bf16* A0 = hB + (size_t)m0 * KP;
      const bf16* A1 = A0 + (size_t)16 * KP;
      v16bf a0 = load_a16x32(A0, KP, 0, lane);
      v16bf a1 = load_a16x32(A1, KP, 0, lane);
#pragma unroll 1
      for (int ks = 0; ks < KSTEPS; ++ks) {
        int k0 = ks * 32;
        v16bf b = load_b_panel(ldsE, VBLK, k0, n0, lane);
        v16bf a0n = a0, a1n = a1;
        if (ks + 1 < KSTEPS) {
          a0n = load_a16x32(A0, KP, k0 + 32, lane);
          a1n = load_a16x32(A1, KP, k0 + 32, lane);
        }
        acc0 = wmma_bf16(a0, b, acc0);
        acc1 = wmma_bf16(a1, b, acc1);
        a0 = a0n; a1 = a1n;
      }
    }
    if (phase == 0) {
      rowstat(acc0, m0);
      rowstat(acc1, m0 + 16);
    } else if (active) {
      store_tile(acc0, m0);
      store_tile(acc1, m0 + 16);
    }
  }
}

__global__ void k_row_reduce(const float* __restrict__ partMax,
                             const float* __restrict__ partSum,
                             float* __restrict__ rowMax,
                             float* __restrict__ rowSum) {
  int m = blockIdx.x * blockDim.x + threadIdx.x;
  if (m >= MROWS) return;
  float Mx = -3.0e38f;
  for (int blk = 0; blk < NVB; ++blk)
    Mx = fmaxf(Mx, partMax[(size_t)blk * MROWS + m]);
  float Sm = 0.0f;
  for (int blk = 0; blk < NVB; ++blk)
    Sm += partSum[(size_t)blk * MROWS + m] *
          __expf(partMax[(size_t)blk * MROWS + m] - Mx);
  rowMax[m] = Mx;
  rowSum[m] = Sm;
}

// Pointer distribution scatter: out[s,b,t,story[b,tpos]] += (1-p_gen)*attn
__global__ void k_scatter(const float* __restrict__ attn,
                          const float* __restrict__ pgen,
                          const int* __restrict__ story,
                          float* __restrict__ dout) {
  long total = (long)MROWS * Tn;
  for (long id = (long)blockIdx.x * blockDim.x + threadIdx.x; id < total;
       id += (long)gridDim.x * blockDim.x) {
    int m = (int)(id / Tn), tpos = (int)(id % Tn);
    float a = attn[(size_t)m * Tn + tpos];
    int b = m / (Sn * STEPSn), rem = m % (Sn * STEPSn);
    int s = rem / STEPSn, t = rem % STEPSn;
    int w = story[(size_t)b * Tn + tpos];
    size_t o = (((size_t)s * Bn + b) * STEPSn + t) * Vn + w;
    atomicAdd(&dout[o], (1.0f - pgen[m]) * a);
  }
}

// ---------------------------------------------------------------------------
// Host orchestration
// ---------------------------------------------------------------------------
extern "C" void kernel_launch(void* const* d_in, const int* in_sizes, int n_in,
                              void* d_out, int out_size, void* d_ws, size_t ws_size,
                              hipStream_t stream) {
  (void)in_sizes; (void)n_in; (void)out_size; (void)ws_size;
  const float* emb        = (const float*)d_in[0];
  const float* Wih        = (const float*)d_in[1];
  const float* Whh        = (const float*)d_in[2];
  const float* bih        = (const float*)d_in[3];
  const float* bhh        = (const float*)d_in[4];
  const float* Wr         = (const float*)d_in[5];
  const float* br         = (const float*)d_in[6];
  const float* Wg         = (const float*)d_in[7];
  const float* bg         = (const float*)d_in[8];
  const float* slot_embs  = (const float*)d_in[9];
  const float* enc_hidden = (const float*)d_in[10];
  const float* enc_out    = (const float*)d_in[11];
  const int*   lens       = (const int*)d_in[12];
  const int*   story      = (const int*)d_in[13];
  const int*   target     = (const int*)d_in[14];
  float* dout = (float*)d_out;

  // carve workspace
  char* ws = (char*)d_ws;
  size_t off = 0;
  auto carve = [&](size_t bytes) {
    void* p = ws + off;
    off += (bytes + 255) & ~(size_t)255;
    return p;
  };
  bf16*  WihP   = (bf16*)carve((size_t)KP * H3 * 2);
  bf16*  WhhP   = (bf16*)carve((size_t)KP * H3 * 2);
  bf16*  EtP    = (bf16*)carve((size_t)KP * Vn * 2);
  bf16*  encBp  = (bf16*)carve((size_t)Bn * TP * Hn * 2);
  bf16*  encTp  = (bf16*)carve((size_t)Bn * KP * Tn * 2);
  bf16*  xB     = (bf16*)carve((size_t)MROWS * KP * 2);
  bf16*  hB     = (bf16*)carve((size_t)MROWS * KP * 2);
  float* xF     = (float*)carve((size_t)MROWS * Hn * 4);
  float* hF     = (float*)carve((size_t)MROWS * Hn * 4);
  float* attn   = (float*)carve((size_t)MROWS * Tn * 4);   // logits then attn
  bf16*  attnB  = (bf16*)carve((size_t)MROWS * TP * 2);
  float* ctxF   = (float*)carve((size_t)MROWS * Hn * 4);
  float* pgen   = (float*)carve((size_t)MROWS * 4);
  float* partMax = (float*)carve((size_t)NVB * MROWS * 4);
  float* partSum = (float*)carve((size_t)NVB * MROWS * 4);
  float* rowMax  = (float*)carve((size_t)MROWS * 4);
  float* rowSum  = (float*)carve((size_t)MROWS * 4);

  const int TB = 256;
  auto nb = [](long n, int tb) { return (int)((n + tb - 1) / tb); };

  // --- prep: bf16 conversion into panelized layouts ---
  k_panelcvt<<<nb((long)KP * H3, TB), TB, 0, stream>>>(WihP, Wih, H3, Hn);
  k_panelcvt<<<nb((long)KP * H3, TB), TB, 0, stream>>>(WhhP, Whh, H3, Hn);
  k_panelcvt<<<nb((long)KP * Vn, TB), TB, 0, stream>>>(EtP, emb, Vn, Hn);
  k_panel_encB<<<nb((long)Bn * TP * Hn, TB), TB, 0, stream>>>(encBp, enc_out);
  k_panel_encT<<<nb((long)Bn * KP * Tn, TB), TB, 0, stream>>>(encTp, enc_out);
  k_build_x<<<nb((long)MROWS * KP, TB), TB, 0, stream>>>(xF, xB, slot_embs, emb,
                                                         target);
  k_zero_bf16<<<nb((long)MROWS * KP, TB), TB, 0, stream>>>(hB, (long)MROWS * KP);

  // --- GRU recurrence: 8 sequential WMMA steps over 480 chains ---
  for (int t = 0; t < STEPSn; ++t)
    k_gru_step<<<PAIRS / 16, TB, 0, stream>>>(xB, hB, hF, WihP, WhhP, bih, bhh,
                                              enc_hidden, t);

  // --- attention logits: per-b GEMM [240 x 400 x 416] ---
  k_tile_gemm<<<Bn * 15, TB, 0, stream>>>(
      hB, KP, (long)240 * KP, encTp, Tn, (long)KP * Tn,
      attn, Tn, (long)240 * Tn, 15, Tn / 16, KP);

  // --- masked softmax over T ---
  k_attn_softmax<<<MROWS, 128, 0, stream>>>(attn, attnB, lens);

  // --- context: per-b GEMM [240 x 400 x 416] ---
  k_tile_gemm<<<Bn * 15, TB, 0, stream>>>(
      attnB, TP, (long)240 * TP, encBp, Hn, (long)TP * Hn,
      ctxF, Hn, (long)240 * Hn, 15, Hn / 16, TP);

  // --- p_gen + gate head (writes gate outputs at tail of d_out) ---
  float* gateOut = dout + (size_t)Sn * Bn * STEPSn * Vn;
  k_pgen_gate<<<nb(MROWS, TB), TB, 0, stream>>>(hF, ctxF, xF, Wr, br, Wg, bg,
                                                pgen, gateOut);

  // --- vocab GEMM + flash softmax (two pass) ---
  size_t shmem = (size_t)KP * VBLK * sizeof(bf16);  // 104KB of 320KB/WGP
  k_vocab<<<NVB, TB, shmem, stream>>>(hB, EtP, pgen, partMax, partSum,
                                      rowMax, rowSum, dout, 0);
  k_row_reduce<<<nb(MROWS, TB), TB, 0, stream>>>(partMax, partSum, rowMax,
                                                 rowSum);
  k_vocab<<<NVB, TB, shmem, stream>>>(hB, EtP, pgen, partMax, partSum,
                                      rowMax, rowSum, dout, 1);

  // --- pointer-distribution scatter ---
  k_scatter<<<nb((long)MROWS * Tn, TB), TB, 0, stream>>>(attn, pgen, story,
                                                         dout);
}
